// MultiheadAttention_75265006895187
// MI455X (gfx1250) — matrix operand order repaired and checked by reading, not compiled, not run on hardware
//
#include <hip/hip_runtime.h>

// MHA for B=2, N=M=2048, D=1024, H=16, HD=64 on gfx1250 (wave32, WMMA).

typedef __attribute__((ext_vector_type(16))) _Float16 v16h;
typedef __attribute__((ext_vector_type(8)))  float    v8f;
typedef __attribute__((ext_vector_type(4)))  int      v4i;

#define BB 2
#define NN 2048
#define MM 2048
#define DD 1024
#define HH 16
#define HD 64

#if __has_builtin(__builtin_amdgcn_global_load_async_to_lds_b128)
#define HAVE_ASYNC_LDS 1
#else
#define HAVE_ASYNC_LDS 0
#endif

#if HAVE_ASYNC_LDS
typedef __attribute__((address_space(1))) v4i gv4i;
typedef __attribute__((address_space(3))) v4i lv4i;
__device__ __forceinline__ void async_cp16(const _Float16* g, _Float16* l) {
  __builtin_amdgcn_global_load_async_to_lds_b128((gv4i*)g, (lv4i*)l, 0, 0);
}
__device__ __forceinline__ void wait_async_le4() {
  asm volatile("s_wait_asynccnt 0x4" ::: "memory");
}
__device__ __forceinline__ void wait_async_0() {
  asm volatile("s_wait_asynccnt 0x0" ::: "memory");
}
#endif

// ---------------------------------------------------------------------------
// Fragment loaders (contiguous k order k = 16*(L/16) + s, same in A and B; wave32)
// ---------------------------------------------------------------------------

// A-matrix 16x32 f16: lane L holds row (L%16); K values:
//   lanes 0-15 : V0..V7 -> K 0..15
//   lanes 16-31: V0..V7 -> K 16..31
// K pairs are contiguous -> 8 dword loads.
__device__ __forceinline__ v16h load_a_frag(const _Float16* X, int rowbase,
                                            int ld, int k0, int lane) {
  int row = rowbase + (lane & 15);
  int kb  = (lane & 16) ? 16 : 0;
  const _Float16* p = X + (size_t)row * ld + k0 + kb;
  union { v16h v; unsigned int u[8]; } a;
#pragma unroll
  for (int pr = 0; pr < 8; ++pr) {
    int k = pr * 2;
    a.u[pr] = *(const unsigned int*)(p + k);
  }
  return a.v;
}

// B-matrix 32x16 f16 where B[k][n] = Bmat[nbase+n][k0+k] (row-major source,
// i.e. multiply by Bmat^T). Lane L holds column n = L%16; lanes 0-15 hold
// K=0..15, lanes 16-31 hold K=16..31 -> one 32B contiguous vector load.
__device__ __forceinline__ v16h load_b_frag_rowmajor(const _Float16* Bmat,
                                                     int nbase, int ld,
                                                     int k0, int lane) {
  int n  = lane & 15;
  int kb = (lane & 16) ? 16 : 0;
  return *(const v16h*)(Bmat + (size_t)(nbase + n) * ld + k0 + kb);
}

// ---------------------------------------------------------------------------
// f32 -> f16 conversion
// ---------------------------------------------------------------------------
__global__ void cvt_f32_f16_kernel(const float* __restrict__ in,
                                   _Float16* __restrict__ out, int n) {
  int i = blockIdx.x * blockDim.x + threadIdx.x;
  if (i < n) out[i] = (_Float16)in[i];
}

// ---------------------------------------------------------------------------
// GEMM: Y[r,c] = sum_k X[r,k]*W[c,k] + bias[c]   (i.e. X @ W^T + b)
// Per-wave 16x64 output tile; block = 8 waves -> 128x64 per block.
// KPAD : zero rows of projected K where pos >= seqM - pad[b] (reference
//        zeroes padded key vectors -> score 0, not -inf).
// TRANSV: write f16 output transposed per head, [b][h][hd][m], so the
//        P@V WMMA B-fragments become contiguous 32B vector loads.
// ---------------------------------------------------------------------------
template <bool OUT_F32, bool KPAD, bool TRANSV>
__global__ void gemm_wmma_kernel(const _Float16* __restrict__ X,
                                 const _Float16* __restrict__ W,
                                 const float* __restrict__ bias,
                                 void* __restrict__ Y,
                                 const int* __restrict__ pad,
                                 int rows, int cols, int kdim, int seqM) {
  int lane = threadIdx.x & 31;
  int wave = threadIdx.x >> 5;
  int rowbase = blockIdx.y * 128 + wave * 16;
  int colbase = blockIdx.x * 64;
  if (rowbase >= rows) return;

  v8f acc[4] = {};
  for (int k0 = 0; k0 < kdim; k0 += 32) {
    if (k0 + 32 < kdim) {  // prefetch next k-chunk (global_prefetch_b8)
      __builtin_prefetch(X + (size_t)(rowbase + (lane & 15)) * kdim + k0 + 32, 0, 1);
      __builtin_prefetch(W + (size_t)(colbase + (lane & 15)) * kdim + k0 + 32, 0, 1);
    }
    v16h a = load_a_frag(X, rowbase, kdim, k0, lane);
#pragma unroll
    for (int nt = 0; nt < 4; ++nt) {
      v16h b = load_b_frag_rowmajor(W, colbase + nt * 16, kdim, k0, lane);
      acc[nt] = __builtin_amdgcn_wmma_f32_16x16x32_f16(
          false, a, false, b, (short)0, acc[nt], false, false);
    }
  }

  int halo = (lane & 16) ? 8 : 0;
#pragma unroll
  for (int nt = 0; nt < 4; ++nt) {
    int col = colbase + nt * 16 + (lane & 15);
    float bv = bias[col];
#pragma unroll
    for (int r = 0; r < 8; ++r) {
      int row = rowbase + r + halo;
      float v = acc[nt][r] + bv;
      if (KPAD) {
        int bidx = row / seqM;
        int pos  = row % seqM;
        if (pos >= seqM - pad[bidx]) v = 0.0f;
      }
      if (OUT_F32) {
        ((float*)Y)[(size_t)row * cols + col] = v;
      } else if (TRANSV) {
        int bidx = row / seqM, pos = row % seqM;
        int h = col >> 6, ch = col & 63;
        ((_Float16*)Y)[(((size_t)bidx * HH + h) * HD + ch) * (size_t)seqM + pos] =
            (_Float16)v;
      } else {
        ((_Float16*)Y)[(size_t)row * cols + col] = (_Float16)v;
      }
    }
  }
}

// ---------------------------------------------------------------------------
// Attention: one wave per (b, h, 16-query tile).
//   pass 1: S = QK^T * 1/8, causal mask (-1e6), staged in 64KB LDS, row max.
//           K tiles double-buffered into LDS with async global->LDS copies.
//   pass 2: softmax in LDS, atomicAdd p/H into mean output
//   pass 3: O = P @ V via WMMA (V pre-transposed -> contiguous B loads)
// ---------------------------------------------------------------------------
__global__ void attn_kernel(const _Float16* __restrict__ Q,
                            const _Float16* __restrict__ K,
                            const _Float16* __restrict__ Vt,
                            const int* __restrict__ maskflag,
                            float* __restrict__ mean,
                            _Float16* __restrict__ O) {
  extern __shared__ _Float16 S[];  // [16][MM] f16 = 64KB
  __shared__ float rowmax_s[16];
#if HAVE_ASYNC_LDS
  __shared__ __align__(32) _Float16 Ktile[2][16 * 64];  // double-buffered
#endif

  int lane  = threadIdx.x;
  int qt    = blockIdx.x;
  int h     = blockIdx.y;
  int b     = blockIdx.z;
  int qbase = qt * 16;
  int halo  = (lane & 16) ? 8 : 0;
  int causal = maskflag[0];

  const _Float16* Qb  = Q + ((size_t)b * NN + qbase) * DD + h * HD;
  const _Float16* Kb  = K + (size_t)b * MM * DD + h * HD;
  const _Float16* Vtb = Vt + ((size_t)b * HH + h) * HD * (size_t)MM;

  // Q tile fragments (16 x 64, two K-chunks of 32), reused across all m-tiles
  v16h aq0 = load_a_frag(Qb, 0, DD, 0, lane);
  v16h aq1 = load_a_frag(Qb, 0, DD, 32, lane);

  float vmax[8];
#pragma unroll
  for (int r = 0; r < 8; ++r) vmax[r] = -3.0e38f;

#if HAVE_ASYNC_LDS
  // Stage one 16x64 f16 K tile (2KB) with 4 async b128 copies per lane.
  auto stage = [&](int mt, int buf) {
#pragma unroll
    for (int i = 0; i < 4; ++i) {
      int idx  = i * 32 + lane;          // 128 chunks of 16B
      int krow = idx >> 3, q = idx & 7;
      async_cp16(Kb + (size_t)(mt * 16 + krow) * DD + q * 8,
                 &Ktile[buf][krow * 64 + q * 8]);
    }
  };
  stage(0, 0);
#endif

  // ---- pass 1: scores ----
  for (int mt = 0; mt < MM / 16; ++mt) {
    int mbase = mt * 16;
    v16h b0, b1;
#if HAVE_ASYNC_LDS
    if (mt + 1 < MM / 16) {
      stage(mt + 1, (mt + 1) & 1);
      wait_async_le4();   // previous tile's 4 copies done (in-order)
    } else {
      wait_async_0();
    }
    b0 = load_b_frag_rowmajor(Ktile[mt & 1], 0, 64, 0, lane);
    b1 = load_b_frag_rowmajor(Ktile[mt & 1], 0, 64, 32, lane);
#else
    b0 = load_b_frag_rowmajor(Kb, mbase, DD, 0, lane);
    b1 = load_b_frag_rowmajor(Kb, mbase, DD, 32, lane);
#endif
    v8f s = {};
    s = __builtin_amdgcn_wmma_f32_16x16x32_f16(false, aq0, false, b0,
                                               (short)0, s, false, false);
    s = __builtin_amdgcn_wmma_f32_16x16x32_f16(false, aq1, false, b1,
                                               (short)0, s, false, false);
    int m = mbase + (lane & 15);
#pragma unroll
    for (int r = 0; r < 8; ++r) {
      float v = s[r] * 0.125f;  // 1/sqrt(64)
      int q = qbase + r + halo;
      if (causal && m > q) v = -1.0e6f;  // f16 store saturates; exp -> 0
      S[(r + halo) * MM + m] = (_Float16)v;
      vmax[r] = fmaxf(vmax[r], v);
    }
  }

  // row-max reduction across the 16 lanes holding each row
#pragma unroll
  for (int r = 0; r < 8; ++r) {
#pragma unroll
    for (int msk = 8; msk >= 1; msk >>= 1)
      vmax[r] = fmaxf(vmax[r], __shfl_xor(vmax[r], msk, 32));
  }
  if ((lane & 15) == 0) {
#pragma unroll
    for (int r = 0; r < 8; ++r) rowmax_s[r + halo] = vmax[r];
  }
  __syncthreads();

  // ---- pass 2: softmax + mean accumulation ----
  int row   = lane & 15;
  int chalf = (lane >> 4) * (MM / 2);
  float mx  = rowmax_s[row];
  float sum = 0.0f;
  for (int c = 0; c < MM / 2; ++c) {
    int cc = chalf + c;
    float p = __expf((float)S[row * MM + cc] - mx);
    S[row * MM + cc] = (_Float16)p;
    sum += p;
  }
  sum += __shfl_xor(sum, 16, 32);
  float inv = 1.0f / sum;
  float* meanrow = mean + ((size_t)b * NN + qbase + row) * MM;
  for (int c = 0; c < MM / 2; ++c) {
    int cc = chalf + c;
    float p = (float)S[row * MM + cc] * inv;
    S[row * MM + cc] = (_Float16)p;
    atomicAdd(meanrow + cc, p * (1.0f / HH));
  }
  __syncthreads();

  // ---- pass 3: O = P @ V  (V transposed: B[k][n] = Vt[n][mc+k]) ----
  v8f acc[4] = {};
  for (int mc = 0; mc < MM; mc += 32) {
    v16h a = load_a_frag(S, 0, MM, mc, lane);
#pragma unroll
    for (int nt = 0; nt < 4; ++nt) {
      v16h bf = load_b_frag_rowmajor(Vtb, nt * 16, MM, mc, lane);
      acc[nt] = __builtin_amdgcn_wmma_f32_16x16x32_f16(
          false, a, false, bf, (short)0, acc[nt], false, false);
    }
  }
  _Float16* Ob = O + ((size_t)b * NN + qbase) * DD + h * HD;
#pragma unroll
  for (int nt = 0; nt < 4; ++nt) {
#pragma unroll
    for (int r = 0; r < 8; ++r)
      Ob[(size_t)(r + halo) * DD + nt * 16 + (lane & 15)] =
          (_Float16)acc[nt][r];
  }
}

// ---------------------------------------------------------------------------
// Host launch
// ---------------------------------------------------------------------------
extern "C" void kernel_launch(void* const* d_in, const int* in_sizes, int n_in,
                              void* d_out, int out_size, void* d_ws,
                              size_t ws_size, hipStream_t stream) {
  const float* query = (const float*)d_in[0];
  const float* key   = (const float*)d_in[1];
  const float* value = (const float*)d_in[2];
  const int*   pad   = (const int*)d_in[3];
  const int*   maskf = (const int*)d_in[4];
  const float* wq_w = (const float*)d_in[5];
  const float* wq_b = (const float*)d_in[6];
  const float* wk_w = (const float*)d_in[7];
  const float* wk_b = (const float*)d_in[8];
  const float* wv_w = (const float*)d_in[9];
  const float* wv_b = (const float*)d_in[10];
  const float* wo_w = (const float*)d_in[11];
  const float* wo_b = (const float*)d_in[12];

  float* out  = (float*)d_out;
  float* mean = out + (size_t)BB * NN * DD;

  const size_t SD = (size_t)BB * NN * DD;  // 4,194,304
  const size_t WD = (size_t)DD * DD;       // 1,048,576

  _Float16* ws  = (_Float16*)d_ws;
  _Float16* xq  = ws;
  _Float16* xk  = xq + SD;
  _Float16* xv  = xk + SD;
  _Float16* wqh = xv + SD;
  _Float16* wkh = wqh + WD;
  _Float16* wvh = wkh + WD;
  _Float16* woh = wvh + WD;
  _Float16* qp  = woh + WD;
  _Float16* kp  = qp + SD;
  _Float16* vp  = kp + SD;  // transposed per head: [b][h][hd][m]
  _Float16* att = vp + SD;

  // 1) convert inputs and weights to f16
  {
    int nS = (int)SD, nW = (int)WD;
    cvt_f32_f16_kernel<<<(nS + 255) / 256, 256, 0, stream>>>(query, xq, nS);
    cvt_f32_f16_kernel<<<(nS + 255) / 256, 256, 0, stream>>>(key, xk, nS);
    cvt_f32_f16_kernel<<<(nS + 255) / 256, 256, 0, stream>>>(value, xv, nS);
    cvt_f32_f16_kernel<<<(nW + 255) / 256, 256, 0, stream>>>(wq_w, wqh, nW);
    cvt_f32_f16_kernel<<<(nW + 255) / 256, 256, 0, stream>>>(wk_w, wkh, nW);
    cvt_f32_f16_kernel<<<(nW + 255) / 256, 256, 0, stream>>>(wv_w, wvh, nW);
    cvt_f32_f16_kernel<<<(nW + 255) / 256, 256, 0, stream>>>(wo_w, woh, nW);
  }

  // 2) projections  (rows = B*N = 4096, cols = D = 1024)
  dim3 ggrid(DD / 64, (BB * NN) / 128);
  gemm_wmma_kernel<false, false, false><<<ggrid, 256, 0, stream>>>(
      xq, wqh, wq_b, qp, nullptr, BB * NN, DD, DD, MM);
  gemm_wmma_kernel<false, true, false><<<ggrid, 256, 0, stream>>>(
      xk, wkh, wk_b, kp, pad, BB * MM, DD, DD, MM);
  gemm_wmma_kernel<false, false, true><<<ggrid, 256, 0, stream>>>(
      xv, wvh, wv_b, vp, nullptr, BB * MM, DD, DD, MM);

  // 3) zero the head-mean output, then attention
  (void)hipMemsetAsync(mean, 0, (size_t)BB * NN * MM * sizeof(float), stream);
  dim3 agrid(NN / 16, HH, BB);
  attn_kernel<<<agrid, 32, 16 * MM * sizeof(_Float16), stream>>>(
      qp, kp, vp, maskf, mean, att);

  // 4) output projection -> f32 d_out
  gemm_wmma_kernel<true, false, false><<<ggrid, 256, 0, stream>>>(
      att, woh, wo_b, out, nullptr, BB * NN, DD, DD, MM);
}
